// GraphGPSModule_85590108275190
// MI455X (gfx1250) — compile-verified, matrix operand order checked by reading, not run on hardware
//
#include <hip/hip_runtime.h>
#include <hip/hip_bf16.h>
#include <stdint.h>

// ---------------- problem constants ----------------
constexpr int N_    = 32768;
constexpr int S_    = 512;
constexpr int DIM_  = 512;
constexpr int HEADS_= 4;
constexpr int E_    = 524288;
constexpr int B_    = N_ / S_;       // 64
constexpr int HDIM_ = DIM_ / HEADS_; // 128
constexpr int BH_   = B_ * HEADS_;   // 256
constexpr float EPS_ = 1e-5f;

// ---------------- vector types ----------------
typedef __attribute__((ext_vector_type(16))) __bf16 bf16x16;
typedef __attribute__((ext_vector_type(8)))  __bf16 bf16x8;
typedef __attribute__((ext_vector_type(8)))  float  f32x8;

// round-to-nearest-even f32 -> bf16 (pure integer ops, no hw cvt dependency)
__device__ __forceinline__ __bf16 tobf(float f) {
  unsigned u = __builtin_bit_cast(unsigned, f);
  unsigned r = u + 0x7FFFu + ((u >> 16) & 1u);
  unsigned short h = (unsigned short)(r >> 16);
  return __builtin_bit_cast(__bf16, h);
}

// Load one 16-bf16 WMMA operand fragment for this lane: two contiguous
// 16-byte chunks at base+off and base+off+16 (CDNA5 16-bit A/B VGPR layout:
// lanes 0-15 hold K {k0..k0+7, k0+16..k0+23}; lanes 16-31 hold {+8, +24}).
__device__ __forceinline__ bf16x16 load_frag(const __bf16* base, int off) {
  bf16x16 r;
  ((bf16x8*)&r)[0] = *(const bf16x8*)(base + off);
  ((bf16x8*)&r)[1] = *(const bf16x8*)(base + off + 16);
  return r;
}

// CDNA5 async global->LDS copy of 16 bytes per lane (tracked on ASYNCcnt).
// lds_off = LDS byte address (low 32 bits of the flat pointer), gaddr = global.
__device__ __forceinline__ void async_b128(unsigned lds_off, const void* gaddr) {
  asm volatile("global_load_async_to_lds_b128 %0, %1, off"
               :: "v"(lds_off), "v"((unsigned long long)(uintptr_t)gaddr)
               : "memory");
}
__device__ __forceinline__ void wait_async0() {
  asm volatile("s_wait_asynccnt 0" ::: "memory");
}
__device__ __forceinline__ unsigned lds_addr32(const void* p) {
  return (unsigned)(uintptr_t)p; // flat LDS aperture: addr[31:0] == LDS offset
}

// ---------------- elementwise / utility kernels ----------------
__global__ void k_zero(float* __restrict__ p, size_t n) {
  size_t i = (size_t)blockIdx.x * blockDim.x + threadIdx.x;
  if (i < n) p[i] = 0.f;
}
__global__ void k_f2b(const float* __restrict__ a, __bf16* __restrict__ o, size_t n) {
  size_t i = (size_t)blockIdx.x * blockDim.x + threadIdx.x;
  if (i < n) o[i] = tobf(a[i]);
}
__global__ void k_add(const float* __restrict__ a, const float* __restrict__ b,
                      float* __restrict__ o, size_t n) {
  size_t i = (size_t)blockIdx.x * blockDim.x + threadIdx.x;
  if (i < n) o[i] = a[i] + b[i];
}
__global__ void k_add_b(const float* __restrict__ a, const float* __restrict__ b,
                        __bf16* __restrict__ o, size_t n) {
  size_t i = (size_t)blockIdx.x * blockDim.x + threadIdx.x;
  if (i < n) o[i] = tobf(a[i] + b[i]);
}
__global__ void k_combine(const float* __restrict__ a, const float* __restrict__ b,
                          float* __restrict__ of, __bf16* __restrict__ ob, size_t n) {
  size_t i = (size_t)blockIdx.x * blockDim.x + threadIdx.x;
  if (i < n) { float v = a[i] + b[i]; of[i] = v; ob[i] = tobf(v); }
}
// W[k][n] (f32, K x Nc row-major) -> Wt[n][k] (bf16) so WMMA B-fragments are
// contiguous per-lane 16B loads.
__global__ void k_wtrans(const float* __restrict__ W, __bf16* __restrict__ Wt,
                         int K, int Nc) {
  size_t i = (size_t)blockIdx.x * blockDim.x + threadIdx.x;
  if (i >= (size_t)K * Nc) return;
  int n = (int)(i / K), k = (int)(i % K);
  Wt[i] = tobf(W[(size_t)k * Nc + n]);
}

// ---------------- edge scatter-add (GIN aggregation) ----------------
// One wave per edge; x is 64MB and L2-resident, so the random gather +
// global_atomic_add_f32 scatter stays in the 192MB L2.
__global__ __launch_bounds__(256) void k_scatter(const int* __restrict__ ei,
                                                 const float* __restrict__ x,
                                                 float* __restrict__ agg) {
  int gid  = blockIdx.x * blockDim.x + threadIdx.x;
  int e    = gid >> 5;
  int lane = gid & 31;
  if (e >= E_) return;
  int src = ei[e];
  int dst = ei[E_ + e];
  const float* xs = x + (size_t)src * DIM_;
  float* ad = agg + (size_t)dst * DIM_;
  #pragma unroll
  for (int f = lane; f < DIM_; f += 32) atomicAdd(&ad[f], xs[f]);
}

// ---------------- generic bf16 WMMA GEMM ----------------
// C[M,Nc] = act( A[M,K](bf16) @ W[K,Nc] + bias ), W given transposed as
// Wt[Nc,K] (bf16). Block = 8 waves; the 64-column weight tile (64 x K bf16,
// contiguous rows of Wt) is async-copied to LDS ONCE per block and shared by
// all 8 waves (8x VMEM reduction vs per-wave global loads). A streams from
// global with double-buffered b128 loads so WMMAs overlap the loads.
// Grid (M/128, Nc/64); dynamic LDS = 64*K*2 bytes.
__global__ __launch_bounds__(256)
void k_gemm(const __bf16* __restrict__ A, const __bf16* __restrict__ Wt,
            const float* __restrict__ bias, float* __restrict__ Cf,
            __bf16* __restrict__ Cb, int M, int K, int Nc, int relu) {
  extern __shared__ __bf16 Bs[]; // [64][K]
  const int tid  = threadIdx.x;
  const int lane = tid & 31;
  const int wave = tid >> 5;
  const int half = lane >> 4;   // which 16-lane group
  const int lid  = lane & 15;
  const int m0 = blockIdx.x * 128 + wave * 16;
  const int n0 = blockIdx.y * 64;

  // ---- stage weight tile: rows n0..n0+63 of Wt are contiguous in memory ----
  {
    const __bf16* gsrc = Wt + (size_t)n0 * K;
    const unsigned lb = lds_addr32(Bs);
    const int total = 64 * K;
    for (int idx = tid * 8; idx < total; idx += 256 * 8)
      async_b128(lb + (unsigned)idx * 2u, gsrc + idx);
    wait_async0();
  }
  __syncthreads();

  f32x8 acc[4] = {};
  const __bf16* arow = A + (size_t)(m0 + lid) * K;
  const int aoff = half * 8;
  // software pipeline the A-fragment global loads one K-step ahead
  bf16x8 alo = *(const bf16x8*)(arow + aoff);
  bf16x8 ahi = *(const bf16x8*)(arow + aoff + 16);
  for (int k0 = 0; k0 < K; k0 += 32) {
    bf16x16 a;
    ((bf16x8*)&a)[0] = alo;
    ((bf16x8*)&a)[1] = ahi;
    if (k0 + 32 < K) {
      __builtin_prefetch(arow + k0 + 96, 0, 1); // global_prefetch next lines
      alo = *(const bf16x8*)(arow + k0 + 32 + aoff);
      ahi = *(const bf16x8*)(arow + k0 + 32 + aoff + 16);
    }
    const int boff = k0 + half * 8;
    bf16x16 b0 = load_frag(Bs + (size_t)(0  + lid) * K, boff);
    bf16x16 b1 = load_frag(Bs + (size_t)(16 + lid) * K, boff);
    bf16x16 b2 = load_frag(Bs + (size_t)(32 + lid) * K, boff);
    bf16x16 b3 = load_frag(Bs + (size_t)(48 + lid) * K, boff);
    acc[0] = __builtin_amdgcn_wmma_f32_16x16x32_bf16(false, a, false, b0, (short)0, acc[0], false, false);
    acc[1] = __builtin_amdgcn_wmma_f32_16x16x32_bf16(false, a, false, b1, (short)0, acc[1], false, false);
    acc[2] = __builtin_amdgcn_wmma_f32_16x16x32_bf16(false, a, false, b2, (short)0, acc[2], false, false);
    acc[3] = __builtin_amdgcn_wmma_f32_16x16x32_bf16(false, a, false, b3, (short)0, acc[3], false, false);
  }
  // C/D layout: VGPR j -> row (half? j+8 : j), col = lane&15 within 16x16 tile
  #pragma unroll
  for (int nf = 0; nf < 4; ++nf) {
    #pragma unroll
    for (int j = 0; j < 8; ++j) {
      int row = m0 + half * 8 + j;
      int col = n0 + nf * 16 + lid;
      float v = acc[nf][j] + bias[col];
      if (relu) v = fmaxf(v, 0.f);
      size_t idx = (size_t)row * Nc + col;
      if (Cf) Cf[idx] = v;
      if (Cb) Cb[idx] = tobf(v);
    }
  }
}

// ---------------- qkv split: scale Q, lay out K row-major, V transposed ----
__global__ void k_qkvsplit(const float* __restrict__ qkv, __bf16* __restrict__ Qh,
                           __bf16* __restrict__ Kh, __bf16* __restrict__ Vt) {
  size_t i = (size_t)blockIdx.x * blockDim.x + threadIdx.x;
  if (i >= (size_t)N_ * DIM_) return;
  int n = (int)(i / DIM_), c = (int)(i % DIM_);
  int h = c / HDIM_, d = c % HDIM_;
  int b = n / S_, s = n % S_;
  int bh = b * HEADS_ + h;
  const float* r = qkv + (size_t)n * (3 * DIM_);
  const float scale = 0.08838834764831845f; // HDIM^-0.5
  Qh[((size_t)bh * S_ + s) * HDIM_ + d] = tobf(r[c] * scale);
  Kh[((size_t)bh * S_ + s) * HDIM_ + d] = tobf(r[DIM_ + c]);
  Vt[((size_t)bh * HDIM_ + d) * S_ + s] = tobf(r[2 * DIM_ + c]);
}

// ---------------- flash attention (per bh, 64-query tile) ----------------
// 4 waves/block, each wave owns 16 query rows; streams 64-key tiles. Per key
// tile the K tile (64x128, contiguous) and V tile (128x64, strided rows) are
// async-copied to LDS once per block and shared by the 4 waves. QK^T and PV
// both run on v_wmma_f32_16x16x32_bf16; P transits C-layout -> LDS -> A-layout.
__global__ __launch_bounds__(128)
void k_attn(const __bf16* __restrict__ Qh, const __bf16* __restrict__ Kh,
            const __bf16* __restrict__ Vt, __bf16* __restrict__ Ob) {
  __shared__ __bf16 Ks[64 * HDIM_];    // key tile: [key 64][d 128]  (16KB)
  __shared__ __bf16 Vs[HDIM_ * 64];    // val tile: [d 128][key 64]  (16KB)
  __shared__ __bf16 Pbuf[4][16][64];   // per-wave P staging          (8KB)
  const int tid  = threadIdx.x;
  const int lane = tid & 31;
  const int wave = tid >> 5;
  const int half = lane >> 4, lid = lane & 15;
  const int bh = blockIdx.x;
  const int q0 = blockIdx.y * 64 + wave * 16;
  const int b = bh >> 2, h = bh & 3;

  const __bf16* Kbase = Kh + (size_t)bh * S_ * HDIM_;
  const __bf16* Vbase = Vt + (size_t)bh * HDIM_ * S_;
  const unsigned ksb = lds_addr32(Ks);
  const unsigned vsb = lds_addr32(Vs);

  const __bf16* qrow = Qh + ((size_t)bh * S_ + q0 + lid) * HDIM_;
  bf16x16 qf[4];
  #pragma unroll
  for (int kc = 0; kc < 4; ++kc) qf[kc] = load_frag(qrow, kc * 32 + half * 8);

  f32x8 O[8] = {};
  float m[8], l[8];
  #pragma unroll
  for (int j = 0; j < 8; ++j) { m[j] = -3.0e38f; l[j] = 0.f; }

  for (int kt = 0; kt < S_ / 64; ++kt) {
    // ---- async stage K/V tiles into LDS (shared by all 4 waves) ----
    {
      const __bf16* kg = Kbase + (size_t)kt * 64 * HDIM_; // contiguous 16KB
      for (int idx = tid * 8; idx < 64 * HDIM_; idx += 128 * 8)
        async_b128(ksb + (unsigned)idx * 2u, kg + idx);
      // V: 128 rows of 64 elems, global row stride S_; 8 chunks per row
      for (int c = tid; c < HDIM_ * 8; c += 128) {
        int d = c >> 3, sub = (c & 7) * 8;
        async_b128(vsb + (unsigned)(d * 64 + sub) * 2u,
                   Vbase + (size_t)d * S_ + kt * 64 + sub);
      }
      wait_async0();
    }
    __syncthreads();

    // ---- S = Q K^T for 64 keys (K fragments from LDS) ----
    f32x8 Sf[4] = {};
    #pragma unroll
    for (int nf = 0; nf < 4; ++nf) {
      const __bf16* krow = Ks + (size_t)(nf * 16 + lid) * HDIM_;
      #pragma unroll
      for (int kc = 0; kc < 4; ++kc) {
        bf16x16 bf = load_frag(krow, kc * 32 + half * 8);
        Sf[nf] = __builtin_amdgcn_wmma_f32_16x16x32_bf16(false, qf[kc], false, bf, (short)0, Sf[nf], false, false);
      }
    }
    // ---- online softmax (rows live in VGPR j across 16-lane halves) ----
    #pragma unroll
    for (int j = 0; j < 8; ++j) {
      float mj = fmaxf(fmaxf(Sf[0][j], Sf[1][j]), fmaxf(Sf[2][j], Sf[3][j]));
      #pragma unroll
      for (int d = 1; d < 16; d <<= 1) mj = fmaxf(mj, __shfl_xor(mj, d, 16));
      float newm = fmaxf(m[j], mj);
      float alpha = __expf(m[j] - newm);
      m[j] = newm;
      l[j] *= alpha;
      #pragma unroll
      for (int hf = 0; hf < 8; ++hf) O[hf][j] *= alpha;
      float s = 0.f;
      #pragma unroll
      for (int nf = 0; nf < 4; ++nf) {
        float p = __expf(Sf[nf][j] - newm);
        Sf[nf][j] = p;
        s += p;
      }
      #pragma unroll
      for (int d = 1; d < 16; d <<= 1) s += __shfl_xor(s, d, 16);
      l[j] += s;
    }
    // ---- stage P (bf16) into LDS row-major, per-wave private tile ----
    #pragma unroll
    for (int nf = 0; nf < 4; ++nf)
      #pragma unroll
      for (int j = 0; j < 8; ++j)
        Pbuf[wave][half * 8 + j][nf * 16 + lid] = tobf(Sf[nf][j]);
    __syncthreads();
    // ---- O += P V (V fragments from LDS) ----
    #pragma unroll
    for (int kk = 0; kk < 2; ++kk) {
      bf16x16 pf = load_frag(&Pbuf[wave][lid][0], kk * 32 + half * 8);
      #pragma unroll
      for (int hf = 0; hf < 8; ++hf) {
        bf16x16 bf = load_frag(Vs + (size_t)(hf * 16 + lid) * 64, kk * 32 + half * 8);
        O[hf] = __builtin_amdgcn_wmma_f32_16x16x32_bf16(false, pf, false, bf, (short)0, O[hf], false, false);
      }
    }
    __syncthreads(); // all reads of Ks/Vs/Pbuf done before next-tile restage
  }
  // ---- normalize + write bf16 output rows (feeds O-proj GEMM) ----
  #pragma unroll
  for (int hf = 0; hf < 8; ++hf)
    #pragma unroll
    for (int j = 0; j < 8; ++j) {
      int srow = q0 + half * 8 + j;
      int col = h * HDIM_ + hf * 16 + lid;
      size_t node = (size_t)b * S_ + srow;
      Ob[node * DIM_ + col] = tobf(O[hf][j] / l[j]);
    }
}

// ---------------- BatchNorm: column stats + apply ----------------
__global__ __launch_bounds__(256)
void k_colstats(const float* __restrict__ Hm, float* __restrict__ sum,
                float* __restrict__ sumsq, int rows_per_blk) {
  int col = blockIdx.x * blockDim.x + threadIdx.x;
  int r0 = blockIdx.y * rows_per_blk;
  float s = 0.f, q = 0.f;
  for (int r = 0; r < rows_per_blk; ++r) {
    float v = Hm[(size_t)(r0 + r) * DIM_ + col];
    s += v; q += v * v;
  }
  atomicAdd(&sum[col], s);
  atomicAdd(&sumsq[col], q);
}
__global__ void k_bnapply(const float* __restrict__ Hm, const float* __restrict__ sum,
                          const float* __restrict__ sumsq, const float* __restrict__ w,
                          const float* __restrict__ bsh, float* __restrict__ out) {
  size_t i = (size_t)blockIdx.x * blockDim.x + threadIdx.x;
  if (i >= (size_t)N_ * DIM_) return;
  int col = (int)(i % DIM_);
  float mean = sum[col] * (1.f / N_);
  float var  = sumsq[col] * (1.f / N_) - mean * mean;
  out[i] = (Hm[i] - mean) * rsqrtf(var + EPS_) * w[col] + bsh[col];
}

// ---------------- host orchestration ----------------
extern "C" void kernel_launch(void* const* d_in, const int* in_sizes, int n_in,
                              void* d_out, int out_size, void* d_ws, size_t ws_size,
                              hipStream_t stream) {
  (void)in_sizes; (void)n_in; (void)out_size; (void)ws_size;
  const float* x        = (const float*)d_in[0];
  const int*   ei       = (const int*)  d_in[1];
  const float* gin_W1   = (const float*)d_in[3];
  const float* gin_b1   = (const float*)d_in[4];
  const float* gin_W2   = (const float*)d_in[5];
  const float* gin_b2   = (const float*)d_in[6];
  const float* Wqkv     = (const float*)d_in[7];
  const float* bqkv     = (const float*)d_in[8];
  const float* Wo       = (const float*)d_in[9];
  const float* bo       = (const float*)d_in[10];
  const float* mlp_W1   = (const float*)d_in[11];
  const float* mlp_b1   = (const float*)d_in[12];
  const float* mlp_W2   = (const float*)d_in[13];
  const float* mlp_b2   = (const float*)d_in[14];
  const float* n1w = (const float*)d_in[15]; const float* n1b = (const float*)d_in[16];
  const float* n2w = (const float*)d_in[17]; const float* n2b = (const float*)d_in[18];
  const float* n3w = (const float*)d_in[19]; const float* n3b = (const float*)d_in[20];
  float* out = (float*)d_out;

  const size_t ND = (size_t)N_ * DIM_;
  char* ws = (char*)d_ws;
  size_t off = 0;
  auto alloc = [&](size_t bytes) { size_t o = off; off = (off + bytes + 255) & ~(size_t)255; return o; };
  float*  agg   = (float*) (ws + alloc(ND * 4));
  __bf16* abf   = (__bf16*)(ws + alloc((size_t)N_ * 1024 * 2));
  __bf16* bbf   = (__bf16*)(ws + alloc((size_t)N_ * 1024 * 2));
  float*  f1    = (float*) (ws + alloc((size_t)N_ * 1536 * 4));
  float*  f2    = (float*) (ws + alloc(ND * 4));
  float*  f3    = (float*) (ws + alloc(ND * 4));
  float*  f4    = (float*) (ws + alloc(ND * 4));
  __bf16* Qh    = (__bf16*)(ws + alloc((size_t)BH_ * S_ * HDIM_ * 2));
  __bf16* Kh    = (__bf16*)(ws + alloc((size_t)BH_ * S_ * HDIM_ * 2));
  __bf16* Vt    = (__bf16*)(ws + alloc((size_t)BH_ * HDIM_ * S_ * 2));
  __bf16* gin1t = (__bf16*)(ws + alloc((size_t)512 * 512 * 2));
  __bf16* gin2t = (__bf16*)(ws + alloc((size_t)512 * 512 * 2));
  __bf16* qkvt  = (__bf16*)(ws + alloc((size_t)1536 * 512 * 2));
  __bf16* ot    = (__bf16*)(ws + alloc((size_t)512 * 512 * 2));
  __bf16* m1t   = (__bf16*)(ws + alloc((size_t)1024 * 512 * 2));
  __bf16* m2t   = (__bf16*)(ws + alloc((size_t)512 * 1024 * 2));
  float*  stats = (float*) (ws + alloc((size_t)6 * DIM_ * 4));
  float* sum0 = stats;             float* sq0 = stats + DIM_;
  float* sum1 = stats + 2 * DIM_;  float* sq1 = stats + 3 * DIM_;
  float* sum2 = stats + 4 * DIM_;  float* sq2 = stats + 5 * DIM_;

  const int T = 256;
  const int gND = (int)((ND + T - 1) / T);
  auto gElems = [&](size_t n) { return (int)((n + T - 1) / T); };
  const size_t lds512  = (size_t)64 * 512 * 2;   // 64KB weight tile, K=512
  const size_t lds1024 = (size_t)64 * 1024 * 2;  // 128KB weight tile, K=1024

  // 0. deterministic init of accumulators
  k_zero<<<gND, T, 0, stream>>>(agg, ND);
  k_zero<<<gElems(6 * DIM_), T, 0, stream>>>(stats, (size_t)6 * DIM_);

  // 1. weight transpose + bf16 convert (tiny, stays in L2)
  k_wtrans<<<gElems(512 * 512), T, 0, stream>>>(gin_W1, gin1t, 512, 512);
  k_wtrans<<<gElems(512 * 512), T, 0, stream>>>(gin_W2, gin2t, 512, 512);
  k_wtrans<<<gElems(512 * 1536), T, 0, stream>>>(Wqkv, qkvt, 512, 1536);
  k_wtrans<<<gElems(512 * 512), T, 0, stream>>>(Wo, ot, 512, 512);
  k_wtrans<<<gElems(512 * 1024), T, 0, stream>>>(mlp_W1, m1t, 512, 1024);
  k_wtrans<<<gElems(1024 * 512), T, 0, stream>>>(mlp_W2, m2t, 1024, 512);

  // 2. attention front-end: qkv = x @ Wqkv + b
  k_f2b<<<gND, T, 0, stream>>>(x, abf, ND);
  k_gemm<<<dim3(N_ / 128, 1536 / 64), T, lds512, stream>>>(abf, qkvt, bqkv, f1, (__bf16*)nullptr, N_, 512, 1536, 0);
  k_qkvsplit<<<gND, T, 0, stream>>>(f1, Qh, Kh, Vt);

  // 3. GIN branch: agg = scatter-add; h = relu((x+agg)W1+b1)W2+b2; BN(h+x)
  k_scatter<<<E_ / 8, T, 0, stream>>>(ei, x, agg);
  k_add_b<<<gND, T, 0, stream>>>(x, agg, bbf, ND);
  k_gemm<<<dim3(N_ / 128, 512 / 64), T, lds512, stream>>>(bbf, gin1t, gin_b1, (float*)nullptr, abf, N_, 512, 512, 1);
  k_gemm<<<dim3(N_ / 128, 512 / 64), T, lds512, stream>>>(abf, gin2t, gin_b2, f1, (__bf16*)nullptr, N_, 512, 512, 0);
  k_add<<<gND, T, 0, stream>>>(f1, x, f2, ND);
  k_colstats<<<dim3(DIM_ / T, N_ / 512), T, 0, stream>>>(f2, sum0, sq0, 512);
  k_bnapply<<<gND, T, 0, stream>>>(f2, sum0, sq0, n1w, n1b, f2);

  // 4. attention core + O-projection; BN(o+x)
  k_attn<<<dim3(BH_, S_ / 64), 128, 0, stream>>>(Qh, Kh, Vt, abf);
  k_gemm<<<dim3(N_ / 128, 512 / 64), T, lds512, stream>>>(abf, ot, bo, f1, (__bf16*)nullptr, N_, 512, 512, 0);
  k_add<<<gND, T, 0, stream>>>(f1, x, f3, ND);
  k_colstats<<<dim3(DIM_ / T, N_ / 512), T, 0, stream>>>(f3, sum1, sq1, 512);
  k_bnapply<<<gND, T, 0, stream>>>(f3, sum1, sq1, n2w, n2b, f3);

  // 5. combine + MLP + BN3 -> d_out
  k_combine<<<gND, T, 0, stream>>>(f2, f3, f4, abf, ND);
  k_gemm<<<dim3(N_ / 128, 1024 / 64), T, lds512, stream>>>(abf, m1t, mlp_b1, (float*)nullptr, bbf, N_, 512, 1024, 1);
  k_gemm<<<dim3(N_ / 128, 512 / 64), T, lds1024, stream>>>(bbf, m2t, mlp_b2, f1, (__bf16*)nullptr, N_, 1024, 512, 0);
  k_add<<<gND, T, 0, stream>>>(f1, f4, f1, ND);
  k_colstats<<<dim3(DIM_ / T, N_ / 512), T, 0, stream>>>(f1, sum2, sq2, 512);
  k_bnapply<<<gND, T, 0, stream>>>(f1, sum2, sq2, n3w, n3b, out);
}
